// Router_with_Balance_9277129360119
// MI455X (gfx1250) — compile-verified, matrix operand order checked by reading, not run on hardware
//
#include <hip/hip_runtime.h>

#define HDIM 4096
#define NEXP 128
#define TOPK 8
#define KSTAGE 128   // K elements staged into LDS per iteration

typedef __bf16 bf16_t;
typedef bf16_t v16bf __attribute__((ext_vector_type(16)));
typedef bf16_t v8bf  __attribute__((ext_vector_type(8)));
typedef bf16_t v4bf  __attribute__((ext_vector_type(4)));
typedef float  v8f   __attribute__((ext_vector_type(8)));
typedef float  v4f   __attribute__((ext_vector_type(4)));

// One-shot pre-pass: W [E,H] f32 -> bf16 into scratch (1 MB, L2-resident).
__global__ __launch_bounds__(256) void w_to_bf16_kernel(
    const float* __restrict__ W, bf16_t* __restrict__ Wb, int n4) {
#if defined(__gfx1250__)
  int i = blockIdx.x * 256 + threadIdx.x;   // float4 index
  if (i < n4) {
    v4f v = *(const v4f*)(W + (size_t)i * 4);
    v4bf o;
    o[0] = (bf16_t)v[0]; o[1] = (bf16_t)v[1];
    o[2] = (bf16_t)v[2]; o[3] = (bf16_t)v[3];
    *(v4bf*)(Wb + (size_t)i * 4) = o;
  }
#endif
}

// One block = 16 tokens x all 128 experts. 8 waves; wave w owns the 16x16
// logit tile for experts [16w, 16w+16). PRECONV=true: Wv is bf16 scratch.
template <bool PRECONV>
__global__ __launch_bounds__(256) void router_topk_kernel(
    const float* __restrict__ x,      // [T, H] f32
    const void*  __restrict__ Wv,     // [E, H] bf16 (PRECONV) or f32
    const float* __restrict__ bias,   // [E]
    float* __restrict__ out_w,        // [T, 8]
    float* __restrict__ out_idx)      // [T, 8] (indices stored as float)
{
#if defined(__gfx1250__)
  __shared__ __align__(16) bf16_t ldsA[16 * KSTAGE];  // staged x, bf16 (4 KB)
  __shared__ float ldsS[16 * NEXP];                   // sigmoid scores (8 KB)
  __shared__ float ldsB[NEXP];                        // router bias

  const int tid  = threadIdx.x;
  const int wave = tid >> 5;
  const int lane = tid & 31;
  const int t0   = blockIdx.x * 16;

  if (tid < NEXP) ldsB[tid] = bias[tid];

  // ISA 16-bit A/B fragment geometry (wave32):
  //  lanes 0-15 : K = {0..7, 16..23};  lanes 16-31 : K = {8..15, 24..31}
  const int  col   = lane & 15;                  // A: token row ; B: expert col
  const int  kbase = (lane < 16) ? 0 : 8;
  const long wrow  = (long)(wave * 16 + col) * HDIM;

  // Cooperative-stage geometry: thread t converts 8 consecutive f32 of one
  // token row: tok = t>>4, kk = (t&15)*8.   (16 tokens x 128 K per stage)
  const int stok = tid >> 4;
  const int skk  = (tid & 15) << 3;
  const float* xrow = x + (long)(t0 + stok) * HDIM + skk;

  v8f acc = {};

  for (int s = 0; s < HDIM / KSTAGE; ++s) {
    const int kg0 = s * KSTAGE;
    __syncthreads();
    {   // stage + convert x tile: 2 global b128 loads, 4 cvt_pk, 1 ds b128 store
      v4f lo = *(const v4f*)(xrow + kg0);
      v4f hi = *(const v4f*)(xrow + kg0 + 4);
      v8bf o;
      o[0] = (bf16_t)lo[0]; o[1] = (bf16_t)lo[1];
      o[2] = (bf16_t)lo[2]; o[3] = (bf16_t)lo[3];
      o[4] = (bf16_t)hi[0]; o[5] = (bf16_t)hi[1];
      o[6] = (bf16_t)hi[2]; o[7] = (bf16_t)hi[3];
      *(v8bf*)(ldsA + stok * KSTAGE + skk) = o;
    }
    if (s + 1 < HDIM / KSTAGE)    // global_prefetch_b8 of the next x stage
      __builtin_prefetch(xrow + kg0 + KSTAGE, 0, 1);
    __syncthreads();

    #pragma unroll
    for (int c = 0; c < KSTAGE / 32; ++c) {
      const int kc = c * 32;
      v16bf a, b;
      {   // A fragment: two ds_load_b128, already bf16
        const bf16_t* ap = ldsA + col * KSTAGE + kc + kbase;
        v8bf alo = *(const v8bf*)(ap);
        v8bf ahi = *(const v8bf*)(ap + 16);
        #pragma unroll
        for (int j = 0; j < 8; ++j) { a[j] = alo[j]; a[8 + j] = ahi[j]; }
      }
      if constexpr (PRECONV) {   // B fragment: two global_load_b128, bf16
        const bf16_t* bp = (const bf16_t*)Wv + wrow + kg0 + kc + kbase;
        v8bf blo = *(const v8bf*)(bp);
        v8bf bhi = *(const v8bf*)(bp + 16);
        #pragma unroll
        for (int j = 0; j < 8; ++j) { b[j] = blo[j]; b[8 + j] = bhi[j]; }
      } else {                   // fallback: f32 W, convert inline
        const float* bp = (const float*)Wv + wrow + kg0 + kc + kbase;
        v4f b0 = *(const v4f*)(bp);
        v4f b1 = *(const v4f*)(bp + 4);
        v4f b2 = *(const v4f*)(bp + 16);
        v4f b3 = *(const v4f*)(bp + 20);
        #pragma unroll
        for (int j = 0; j < 4; ++j) {
          b[j]      = (bf16_t)b0[j];
          b[4 + j]  = (bf16_t)b1[j];
          b[8 + j]  = (bf16_t)b2[j];
          b[12 + j] = (bf16_t)b3[j];
        }
      }
      acc = __builtin_amdgcn_wmma_f32_16x16x32_bf16(
          false, a, false, b, (short)0, acc, false, false);
    }
  }

  // C/D layout: lane holds column n = 16*wave + (lane&15),
  // rows M = (lane>>4)*8 + r. Scatter sigmoid(logit) to LDS.
  {
    const int n  = wave * 16 + col;
    const int m0 = (lane >> 4) * 8;
    #pragma unroll
    for (int r = 0; r < 8; ++r) {
      float sc = 1.0f / (1.0f + __expf(-acc[r]));
      ldsS[(m0 + r) * NEXP + n] = sc;
    }
  }
  __syncthreads();

  // Top-8 per token: 16 threads, serial argmax (strict > == jax tie-break).
  if (tid < 16) {
    const int    tok = t0 + tid;
    const float* row = ldsS + tid * NEXP;
    unsigned long long sel0 = 0ull, sel1 = 0ull;
    float wv[TOPK];
    int   iv[TOPK];
    float l1 = 0.0f;
    for (int p = 0; p < TOPK; ++p) {
      float best = -3.0e38f;
      int   bi   = 0;
      for (int e = 0; e < NEXP; ++e) {
        bool taken = (e < 64) ? ((sel0 >> e) & 1ull) : ((sel1 >> (e - 64)) & 1ull);
        float bv = row[e] + ldsB[e];           // bias steers selection only
        if (!taken && bv > best) { best = bv; bi = e; }
      }
      if (bi < 64) sel0 |= 1ull << bi; else sel1 |= 1ull << (bi - 64);
      iv[p] = bi;
      wv[p] = row[bi];                          // gather raw sigmoid score
      l1   += fabsf(wv[p]);
    }
    const float denom = fmaxf(l1, 1e-12f);
    #pragma unroll
    for (int p = 0; p < TOPK; ++p) {
      out_w[(long)tok * TOPK + p]   = wv[p] / denom;
      out_idx[(long)tok * TOPK + p] = (float)iv[p];
    }
  }
#endif
}

extern "C" void kernel_launch(void* const* d_in, const int* in_sizes, int n_in,
                              void* d_out, int out_size, void* d_ws, size_t ws_size,
                              hipStream_t stream) {
  (void)n_in; (void)out_size;
  const float* x    = (const float*)d_in[0];
  const float* W    = (const float*)d_in[1];
  const float* bias = (const float*)d_in[2];
  // d_in[3] is k (== 8), baked in as TOPK.

  const int T = in_sizes[0] / HDIM;                    // 32768
  float* out_w   = (float*)d_out;                      // [T, 8] weights
  float* out_idx = (float*)d_out + (size_t)T * TOPK;   // [T, 8] indices

  const size_t wbytes = (size_t)NEXP * HDIM * sizeof(bf16_t);  // 1 MB
  dim3 grid(T / 16), block(256);

  if (ws_size >= wbytes) {
    bf16_t* Wb = (bf16_t*)d_ws;
    const int n4 = NEXP * HDIM / 4;
    w_to_bf16_kernel<<<dim3((n4 + 255) / 256), block, 0, stream>>>(W, Wb, n4);
    router_topk_kernel<true><<<grid, block, 0, stream>>>(x, Wb, bias, out_w, out_idx);
  } else {
    router_topk_kernel<false><<<grid, block, 0, stream>>>(x, W, bias, out_w, out_idx);
  }
}
